// EquivariantConvolutionBlock_15333033247056
// MI455X (gfx1250) — matrix operand order repaired and checked by reading, not compiled, not run on hardware
//
#include <hip/hip_runtime.h>
#include <stdint.h>

typedef __attribute__((ext_vector_type(16))) __bf16 v16bf;
typedef __attribute__((ext_vector_type(8)))  float  v8f;

#define NPTS   300000
#define NTILES (NPTS / 16)          // 18750, exact
#define SPAN   2
#define GP     132                  // 128 + 2*SPAN
#define GP3    (GP * GP * GP)       // 2299968
#define NOFF   125
#define CHUNK  5                    // offsets staged in LDS per round
#define TILES_PER_BLOCK 16          // 8 waves x 2 tiles each

// ---- workspace layout (bytes) ----
static const size_t WS_BKER  = 0;              // 125*5*32*16 u16 = 640 KB
static const size_t WS_FEATB = 1u  << 20;      // 300000*32 u16   = 19.2 MB
static const size_t WS_VOL   = 22u << 20;      // 132^3 i32       = 9.2 MB
static const size_t WS_STATS = 33u << 20;      // 128 f32
static const size_t WS_CONV  = 35u << 20;      // 300000*80 f32   = 96 MB

__device__ __forceinline__ uint16_t f2bf(float f) {
  uint32_t u = __float_as_uint(f);
  uint32_t r = u + 0x7FFFu + ((u >> 16) & 1u);   // round-to-nearest-even
  return (uint16_t)(r >> 16);
}

// ---------------- kernel tensor build (125 x 32 x 80 -> bf16, B-matrix lane layout) ---------
__global__ void k_build_ker(const float* __restrict__ tp, uint16_t* __restrict__ bker) {
  int m  = blockIdx.x;                       // offset index, lat = (m%5-2,(m/5)%5-2,m/25-2)
  int o0 = m % 5 - 2, o1 = (m / 5) % 5 - 2, o2 = m / 25 - 2;
  float d = sqrtf((float)(o0 * o0 + o1 * o1 + o2 * o2));
  const float vals[3] = {0.625f, 1.25f, 1.875f};
  float emb[3];
#pragma unroll
  for (int j = 0; j < 3; ++j) {
    float diff = (d - vals[j]) * (1.0f / 0.625f);
    float t1 = diff + 1.0f, t2 = 1.0f - diff;
    float f1 = t1 > 0.f ? __expf(-1.0f / t1) : 0.f;
    float f2 = t2 > 0.f ? __expf(-1.0f / t2) : 0.f;
    emb[j] = (1.14136f * 7.3890560989f) * f1 * f2;
  }
  float invd = d > 0.f ? 1.0f / d : 0.f;
  float y1[3] = {1.73205081f * o0 * invd, 1.73205081f * o1 * invd, 1.73205081f * o2 * invd};

  for (int t = threadIdx.x; t < 32 * 80; t += blockDim.x) {
    int r = t / 80, c = t % 80;
    float val;
    // w[col] = (emb . tp[:,col]) / 125
    #define WCOL(col) ((emb[0]*tp[(col)] + emb[1]*tp[768+(col)] + emb[2]*tp[1536+(col)]) * (1.0f/125.0f))
    if (r < 8) {
      int u = r;
      if (c < 16)       val = 0.25f * WCOL(0 * 128 + u * 16 + c);
      else if (c < 32)  val = 0.25f * WCOL(2 * 128 + u * 16 + (c - 16));
      else { int q = c - 32, w = q / 3, k = q % 3;
             val = 0.25f * y1[k] * WCOL(4 * 128 + u * 16 + w); }
    } else {
      int rr = r - 8, u = rr / 3, i = rr % 3;
      if (c < 16)       val = 0.14433757f * y1[i] * WCOL(1 * 128 + u * 16 + c);
      else if (c < 32)  val = 0.14433757f * y1[i] * WCOL(3 * 128 + u * 16 + (c - 16));
      else { int q = c - 32, w = q / 3, k = q % 3;
             val = (i == k) ? 0.25f * WCOL(5 * 128 + u * 16 + w) : 0.f; }
    }
    #undef WCOL
    // B (32x16 bf16), contiguous k order matching A: lanes 0-15 hold K=0..15, lanes 16-31 hold K=16..31
    int tt = c >> 4, n = c & 15, k = r;
    int lane = n + ((k >= 16) ? 16 : 0);
    int e    = k & 15;
    bker[(((size_t)m * 5 + tt) * 32 + lane) * 16 + e] = f2bf(val);
  }
}

// ---------------- prep kernels ----------------
__global__ void k_cvt_feats(const float* __restrict__ f, uint16_t* __restrict__ fb) {
  int i = blockIdx.x * 256 + threadIdx.x;
  if (i < NPTS * 32) fb[i] = f2bf(f[i]);
}
__global__ void k_vol_init(int* __restrict__ vol) {
  int i = blockIdx.x * 256 + threadIdx.x;
  if (i < GP3) vol[i] = -1;
}
__global__ void k_scatter(const int* __restrict__ coords, int* __restrict__ vol) {
  int i = blockIdx.x * 256 + threadIdx.x;
  if (i < NPTS) {
    int c0 = coords[i * 3 + 0] + SPAN;
    int c1 = coords[i * 3 + 1] + SPAN;
    int c2 = coords[i * 3 + 2] + SPAN;
    vol[(c0 * GP + c1) * GP + c2] = i;
  }
}

// ---------------- WMMA gather-conv: 2 point-tiles (M=32) per wave ----------------
__global__ void __launch_bounds__(256) k_conv(const uint16_t* __restrict__ featb,
                                              const int* __restrict__ vol,
                                              const int* __restrict__ coords,
                                              const uint16_t* __restrict__ bker,
                                              float* __restrict__ conv) {
  __shared__ uint4 ldsB[CHUNK * 5 * 32 * 2];   // 25.6 KB
  int tid  = threadIdx.x;
  int wave = tid >> 5;
  int lane = tid & 31;
  int tileA = blockIdx.x * TILES_PER_BLOCK + wave * 2;
  int tileB = tileA + 1;
  bool validA = (tileA < NTILES), validB = (tileB < NTILES);
  if (!validA) tileA = NTILES - 1;             // clamp: keep barriers + full EXEC
  if (!validB) tileB = NTILES - 1;
  int half = (lane >> 4) & 1;                  // K-half this lane supplies
  int mrow = lane & 15;

  int rowA = tileA * 16 + mrow;
  int rowB = tileB * 16 + mrow;
  int aLin, bLin;
  {
    int c0 = coords[rowA * 3 + 0] + SPAN;
    int c1 = coords[rowA * 3 + 1] + SPAN;
    int c2 = coords[rowA * 3 + 2] + SPAN;
    aLin = (c0 * GP + c1) * GP + c2;
    c0 = coords[rowB * 3 + 0] + SPAN;
    c1 = coords[rowB * 3 + 1] + SPAN;
    c2 = coords[rowB * 3 + 2] + SPAN;
    bLin = (c0 * GP + c1) * GP + c2;
  }

  v8f z;
#pragma unroll
  for (int j = 0; j < 8; ++j) z[j] = 0.f;
  v8f accA[5] = {z, z, z, z, z};
  v8f accB[5] = {z, z, z, z, z};

  const uint4* bvec = (const uint4*)bker;
  for (int ch = 0; ch < NOFF / CHUNK; ++ch) {
    __syncthreads();
    for (int i = tid; i < CHUNK * 5 * 32 * 2; i += 256)
      ldsB[i] = bvec[ch * (CHUNK * 5 * 32 * 2) + i];
    __syncthreads();
#pragma unroll
    for (int mm = 0; mm < CHUNK; ++mm) {
      int m  = ch * CHUNK + mm;
      int o0 = m % 5 - 2, o1 = (m / 5) % 5 - 2, o2 = m / 25 - 2;
      int delta = (o0 * GP + o1) * GP + o2;
      int idxA = vol[aLin + delta];
      int idxB = vol[bLin + delta];
      int safeA = (idxA < 0) ? 0 : idxA;
      int safeB = (idxB < 0) ? 0 : idxB;
      // A, contiguous k order (B packed the same): lanes 0-15 -> K 0..15 ; lanes 16-31 -> K 16..31
      const uint4* fvA = (const uint4*)(featb + (size_t)safeA * 32 + half * 16);
      const uint4* fvB = (const uint4*)(featb + (size_t)safeB * 32 + half * 16);
      uint4 a0 = fvA[0], a1 = fvA[1];
      uint4 b0 = fvB[0], b1 = fvB[1];
      if (idxA < 0) { a0.x = a0.y = a0.z = a0.w = 0u; a1.x = a1.y = a1.z = a1.w = 0u; }
      if (idxB < 0) { b0.x = b0.y = b0.z = b0.w = 0u; b1.x = b1.y = b1.z = b1.w = 0u; }
      union { v16bf v; uint4 q[2]; } A0, A1;
      A0.q[0] = a0; A0.q[1] = a1;
      A1.q[0] = b0; A1.q[1] = b1;
#pragma unroll
      for (int t = 0; t < 5; ++t) {
        union { v16bf v; uint4 q[2]; } B;
        B.q[0] = ldsB[((mm * 5 + t) * 32 + lane) * 2 + 0];
        B.q[1] = ldsB[((mm * 5 + t) * 32 + lane) * 2 + 1];
        accA[t] = __builtin_amdgcn_wmma_f32_16x16x32_bf16(
            false, A0.v, false, B.v, (short)0, accA[t], false, false);
        accB[t] = __builtin_amdgcn_wmma_f32_16x16x32_bf16(
            false, A1.v, false, B.v, (short)0, accB[t], false, false);
      }
    }
  }
  // D layout: element j -> row (tile*16 + j + half*8), column lane&15
  int n = lane & 15;
  if (validA) {
    int mrow0 = tileA * 16 + half * 8;
#pragma unroll
    for (int t = 0; t < 5; ++t)
#pragma unroll
      for (int j = 0; j < 8; ++j)
        conv[(size_t)(mrow0 + j) * 80 + t * 16 + n] = accA[t][j];
  }
  if (validB) {
    int mrow0 = tileB * 16 + half * 8;
#pragma unroll
    for (int t = 0; t < 5; ++t)
#pragma unroll
      for (int j = 0; j < 8; ++j)
        conv[(size_t)(mrow0 + j) * 80 + t * 16 + n] = accB[t][j];
  }
}

// ---------------- epilogue ----------------
__device__ __forceinline__ void compute_x(int row, const float* __restrict__ feats,
                                          const float* __restrict__ conv,
                                          const float* __restrict__ Ws1,
                                          const float* __restrict__ Ws2,
                                          const float* __restrict__ Wv,
                                          float* x) {
  const float inv = 0.35355339059327373f;   // 1/sqrt(8)
  float xs[8], xv[24];
#pragma unroll
  for (int u = 0; u < 8; ++u)  xs[u] = feats[(size_t)row * 32 + u];
#pragma unroll
  for (int i = 0; i < 24; ++i) xv[i] = feats[(size_t)row * 32 + 8 + i];
#pragma unroll
  for (int c = 0; c < 16; ++c) {
    float a = 0.f, b = 0.f;
#pragma unroll
    for (int u = 0; u < 8; ++u) { a += xs[u] * Ws1[u * 16 + c]; b += xs[u] * Ws2[u * 16 + c]; }
    x[c]      = a * inv + conv[(size_t)row * 80 + c];
    x[16 + c] = b * inv + conv[(size_t)row * 80 + 16 + c];
  }
#pragma unroll
  for (int w = 0; w < 16; ++w)
#pragma unroll
    for (int i = 0; i < 3; ++i) {
      float a = 0.f;
#pragma unroll
      for (int u = 0; u < 8; ++u) a += xv[u * 3 + i] * Wv[u * 16 + w];
      x[32 + w * 3 + i] = a * inv + conv[(size_t)row * 80 + 32 + w * 3 + i];
    }
}

__global__ void k_zero_stats(float* stats) {
  if (threadIdx.x < 128) stats[threadIdx.x] = 0.f;
}

__global__ void __launch_bounds__(256) k_stats(const float* __restrict__ feats,
                                               const float* __restrict__ conv,
                                               const float* __restrict__ Ws1,
                                               const float* __restrict__ Ws2,
                                               const float* __restrict__ Wv,
                                               float* __restrict__ stats) {
  __shared__ float red[48];
  int tid = threadIdx.x;
  if (tid < 48) red[tid] = 0.f;
  __syncthreads();
  int row = blockIdx.x * 256 + tid;
  if (row < NPTS) {
    float x[80];
    compute_x(row, feats, conv, Ws1, Ws2, Wv, x);
#pragma unroll
    for (int c = 0; c < 16; ++c) {
      float s = x[c] / (1.f + __expf(-x[c]));
      atomicAdd(&red[c], s);
      atomicAdd(&red[16 + c], s * s);
    }
#pragma unroll
    for (int w = 0; w < 16; ++w) {
      float g  = 1.f / (1.f + __expf(-x[16 + w]));
      float v0 = x[32 + w * 3 + 0] * g;
      float v1 = x[32 + w * 3 + 1] * g;
      float v2 = x[32 + w * 3 + 2] * g;
      atomicAdd(&red[32 + w], (v0 * v0 + v1 * v1 + v2 * v2) * (1.f / 3.f));
    }
  }
  __syncthreads();
  if (tid < 48) atomicAdd(&stats[tid], red[tid]);
}

__global__ void k_finalize(const float* __restrict__ bn_w, const float* __restrict__ bn_b,
                           float* __restrict__ stats) {
  int c = threadIdx.x;
  if (c < 16) {
    float mu  = stats[c] * (1.0f / NPTS);
    float var = stats[16 + c] * (1.0f / NPTS) - mu * mu;
    float vn  = stats[32 + c] * (1.0f / NPTS);
    stats[64 + c]  = mu;
    stats[80 + c]  = rsqrtf(var + 1e-5f) * bn_w[c];
    stats[96 + c]  = bn_b[c];
    stats[112 + c] = rsqrtf(vn + 1e-5f) * bn_w[16 + c];
  }
}

__global__ void __launch_bounds__(256) k_apply(const float* __restrict__ feats,
                                               const float* __restrict__ conv,
                                               const float* __restrict__ Ws1,
                                               const float* __restrict__ Ws2,
                                               const float* __restrict__ Wv,
                                               const float* __restrict__ stats,
                                               float* __restrict__ out) {
  int row = blockIdx.x * 256 + threadIdx.x;
  if (row >= NPTS) return;
  float x[80];
  compute_x(row, feats, conv, Ws1, Ws2, Wv, x);
#pragma unroll
  for (int c = 0; c < 16; ++c) {
    float s = x[c] / (1.f + __expf(-x[c]));
    out[(size_t)row * 64 + c] = (s - stats[64 + c]) * stats[80 + c] + stats[96 + c];
  }
#pragma unroll
  for (int w = 0; w < 16; ++w) {
    float g = 1.f / (1.f + __expf(-x[16 + w]));
#pragma unroll
    for (int i = 0; i < 3; ++i)
      out[(size_t)row * 64 + 16 + w * 3 + i] = x[32 + w * 3 + i] * g * stats[112 + w];
  }
}

extern "C" void kernel_launch(void* const* d_in, const int* in_sizes, int n_in,
                              void* d_out, int out_size, void* d_ws, size_t ws_size,
                              hipStream_t stream) {
  const float* feats  = (const float*)d_in[0];
  const int*   coords = (const int*)  d_in[1];
  const float* Ws1    = (const float*)d_in[2];
  const float* Ws2    = (const float*)d_in[3];
  const float* Wv     = (const float*)d_in[4];
  const float* tp     = (const float*)d_in[5];
  const float* bn_w   = (const float*)d_in[6];
  const float* bn_b   = (const float*)d_in[7];
  float* out = (float*)d_out;

  char* ws = (char*)d_ws;
  uint16_t* bker  = (uint16_t*)(ws + WS_BKER);
  uint16_t* featb = (uint16_t*)(ws + WS_FEATB);
  int*      vol   = (int*)     (ws + WS_VOL);
  float*    stats = (float*)   (ws + WS_STATS);
  float*    conv  = (float*)   (ws + WS_CONV);

  k_build_ker<<<NOFF, 256, 0, stream>>>(tp, bker);
  k_cvt_feats<<<(NPTS * 32 + 255) / 256, 256, 0, stream>>>(feats, featb);
  k_vol_init <<<(GP3 + 255) / 256, 256, 0, stream>>>(vol);
  k_scatter  <<<(NPTS + 255) / 256, 256, 0, stream>>>(coords, vol);
  k_conv     <<<(NTILES + TILES_PER_BLOCK - 1) / TILES_PER_BLOCK, 256, 0, stream>>>(
      featb, vol, coords, bker, conv);
  k_zero_stats<<<1, 128, 0, stream>>>(stats);
  k_stats    <<<(NPTS + 255) / 256, 256, 0, stream>>>(feats, conv, Ws1, Ws2, Wv, stats);
  k_finalize <<<1, 32, 0, stream>>>(bn_w, bn_b, stats);
  k_apply    <<<(NPTS + 255) / 256, 256, 0, stream>>>(feats, conv, Ws1, Ws2, Wv, stats, out);
}
